// LearnableLogOptimalTransport_28905129902248
// MI455X (gfx1250) — matrix operand order repaired and checked by reading, not compiled, not run on hardware
//
#include <hip/hip_runtime.h>
#include <hip/hip_bf16.h>
#include <stdint.h>

// ---------------------------------------------------------------------------
// Log-domain Sinkhorn (learnable OT) for MI455X / gfx1250.
//
// Reformulation: E = exp(scores) (f16, 67 MB, L2-resident in the 192MB L2),
// so each logsumexp pass is a GEMV:  rowsum = E * exp(v - vmax)  (+ dustbin).
// Row pass uses V_WMMA_F32_16X16X32_F16 with the wv vector broadcast across
// the 16 B-matrix columns. Column pass is a coalesced per-thread reduction.
// E is staged inside d_out (only the final epilogue overwrites d_out);
// u / v / per-batch stats (~263 KB) live in d_ws.
// ---------------------------------------------------------------------------

typedef __attribute__((ext_vector_type(16))) _Float16 v16h;
typedef __attribute__((ext_vector_type(8)))  _Float16 v8h;
typedef __attribute__((ext_vector_type(4)))  _Float16 v4h;
typedef __attribute__((ext_vector_type(8)))  float    v8f;

#define OT_B   32
#define OT_M   1024
#define OT_N   1024
#define OT_MN  (OT_M * OT_N)   // 2^20
#define OT_P   1025            // padded (dustbin) dim
#define OT_ITER 20
#define NINF   (-1.0e6f)

// ---------------- block reductions (wave32, 8 waves / 256 threads) ----------
__device__ __forceinline__ float warpMax(float v) {
#pragma unroll
  for (int o = 16; o > 0; o >>= 1) v = fmaxf(v, __shfl_xor(v, o, 32));
  return v;
}
__device__ __forceinline__ float warpSum(float v) {
#pragma unroll
  for (int o = 16; o > 0; o >>= 1) v += __shfl_xor(v, o, 32);
  return v;
}
__device__ float blockMax(float v, float* s) {
  int lane = threadIdx.x & 31, wid = threadIdx.x >> 5;
  v = warpMax(v);
  if (lane == 0) s[wid] = v;
  __syncthreads();
  float r = (threadIdx.x < 8) ? s[threadIdx.x] : -3.0e38f;
  if (wid == 0) { r = warpMax(r); if (lane == 0) s[0] = r; }
  __syncthreads();
  r = s[0];
  __syncthreads();
  return r;
}
__device__ float blockSum(float v, float* s) {
  int lane = threadIdx.x & 31, wid = threadIdx.x >> 5;
  v = warpSum(v);
  if (lane == 0) s[wid] = v;
  __syncthreads();
  float r = (threadIdx.x < 8) ? s[threadIdx.x] : 0.0f;
  if (wid == 0) { r = warpSum(r); if (lane == 0) s[0] = r; }
  __syncthreads();
  r = s[0];
  __syncthreads();
  return r;
}

// ---------------- K1: per-batch stats + u/v init ---------------------------
__global__ __launch_bounds__(256) void ot_stats_init(
    const unsigned char* __restrict__ rm, const unsigned char* __restrict__ cm,
    const float* __restrict__ alpha, float* __restrict__ stats,
    float* __restrict__ u, float* __restrict__ v) {
  __shared__ float sred[8];
  int b = blockIdx.x, t = threadIdx.x;
  float lr = 0.f, lc = 0.f;
  for (int i = t; i < OT_M; i += 256) lr += rm[b * OT_M + i] ? 1.f : 0.f;
  for (int i = t; i < OT_N; i += 256) lc += cm[b * OT_N + i] ? 1.f : 0.f;
  float nvr = blockSum(lr, sred);
  float nvc = blockSum(lc, sred);
  if (t == 0) {
    float norm = -__logf(nvr + nvc);
    stats[b]      = norm;               // log_mu / log_nu body value
    stats[32 + b] = __logf(nvc) + norm; // log_mu dustbin
    stats[64 + b] = __logf(nvr) + norm; // log_nu dustbin
    if (b == 0) { stats[96] = __expf(alpha[0]); stats[97] = alpha[0]; }
  }
  for (int i = t; i < OT_P; i += 256) {
    u[b * OT_P + i] = 0.f;
    v[b * OT_P + i] = 0.f;
  }
}

// ---------------- K2: E = exp(scores) in f16, masked-to-zero ---------------
__global__ __launch_bounds__(256) void ot_exp(
    const float* __restrict__ scores, const unsigned char* __restrict__ rm,
    const unsigned char* __restrict__ cm, _Float16* __restrict__ E) {
  size_t e = ((size_t)blockIdx.x * 256 + threadIdx.x) * 4;
  int b = (int)(e >> 20);
  int rrem = (int)(e & (OT_MN - 1));
  int i = rrem >> 10, j = rrem & (OT_N - 1);
  const float4 s4 = *(const float4*)(scores + e);
  bool r = rm[b * OT_M + i] != 0;
  const uchar4 c4 = *(const uchar4*)(cm + b * OT_N + j);
  v4h o;
  o[0] = (r && c4.x) ? (_Float16)__expf(s4.x) : (_Float16)0.f;
  o[1] = (r && c4.y) ? (_Float16)__expf(s4.y) : (_Float16)0.f;
  o[2] = (r && c4.z) ? (_Float16)__expf(s4.z) : (_Float16)0.f;
  o[3] = (r && c4.w) ? (_Float16)__expf(s4.w) : (_Float16)0.f;
  *(v4h*)(E + e) = o;
}

// ---------------- K3: u-pass (row sums via WMMA GEMV) ----------------------
// grid (B, 9): y<8 -> 128 rows each (8 waves x 16 rows); y==8 -> dustbin row.
__global__ __launch_bounds__(256) void ot_pass_u(
    const _Float16* __restrict__ E, const float* __restrict__ v,
    float* __restrict__ u, const float* __restrict__ stats,
    const unsigned char* __restrict__ rm) {
  __shared__ __align__(16) _Float16 s_wv[OT_N];
  __shared__ float sred[8];
  int b = blockIdx.x, t = threadIdx.x;
  const float* vb = v + b * OT_P;

  float lm = -3.0e38f;
  for (int j = t; j < OT_P; j += 256) lm = fmaxf(lm, vb[j]);
  float vmax = blockMax(lm, sred);
  for (int j = t; j < OT_N; j += 256) s_wv[j] = (_Float16)__expf(vb[j] - vmax);
  __syncthreads();

  float expalpha = stats[96];
  float norm = stats[b];
  float wvDust = __expf(vb[OT_N] - vmax);

  if (blockIdx.y == 8) {  // dustbin row: u[M] = log_mu_dust - (alpha + lse(v))
    float ls = 0.f;
    for (int j = t; j < OT_P; j += 256) ls += __expf(vb[j] - vmax);
    float S = blockSum(ls, sred);
    if (t == 0) {
      float alpha = stats[97];
      u[b * OT_P + OT_N] = stats[32 + b] - alpha - __logf(fmaxf(S, 1e-38f)) - vmax;
    }
    return;
  }

  int wave = t >> 5, lane = t & 31;
  int h = lane >> 4;                       // lane half
  int rowBase = blockIdx.y * 128 + wave * 16;
  int row = rowBase + (lane & 15);
  const _Float16* Erow = E + ((size_t)b << 20) + ((size_t)row << 10);

  v8f acc = {};
  for (int k = 0; k < OT_N; k += 32) {
    // A (16x32 f16): half h holds K = k+8h..+7 and k+16+8h..+7 (ISA layout)
    v8h a0 = *(const v8h*)(Erow + k + 8 * h);
    v8h a1 = *(const v8h*)(Erow + k + 16 + 8 * h);
    // B (32x16 f16, all 16 columns == wv): half h holds K = k+8h..+7 and k+16+8h..+7
    v8h b0 = *(const v8h*)(s_wv + k + 8 * h);
    v8h b1 = *(const v8h*)(s_wv + k + 16 + 8 * h);
    v16h A, Bm;
#pragma unroll
    for (int x = 0; x < 8; ++x) {
      A[x] = a0[x]; A[x + 8] = a1[x];
      Bm[x] = b0[x]; Bm[x + 8] = b1[x];
    }
    acc = __builtin_amdgcn_wmma_f32_16x16x32_f16(
        false, A, false, Bm, (short)0, acc, false, false);
  }

  // D layout: VGPR r -> (row r, col=lane) for lanes 0-15, (row r+8) for 16-31.
  // All columns identical (broadcast B) -> lanes 0 and 16 hold the results.
  if ((lane & 15) == 0) {
#pragma unroll
    for (int r = 0; r < 8; ++r) {
      int rr = rowBase + r + 8 * h;
      float total = acc[r] + expalpha * wvDust;
      float lmu = rm[b * OT_M + rr] ? norm : NINF;
      u[b * OT_P + rr] = lmu - __logf(fmaxf(total, 1e-38f)) - vmax;
    }
  }
}

// ---------------- K4: v-pass (column sums, coalesced) ----------------------
// grid (B, 5): y<4 -> 256 columns each; y==4 -> dustbin column.
__global__ __launch_bounds__(256) void ot_pass_v(
    const _Float16* __restrict__ E, const float* __restrict__ u,
    float* __restrict__ v, const float* __restrict__ stats,
    const unsigned char* __restrict__ cm) {
  __shared__ float s_wu[OT_P];
  __shared__ float sred[8];
  int b = blockIdx.x, t = threadIdx.x;
  const float* ub = u + b * OT_P;

  float lm = -3.0e38f;
  for (int i = t; i < OT_P; i += 256) lm = fmaxf(lm, ub[i]);
  float umax = blockMax(lm, sred);
  for (int i = t; i < OT_P; i += 256) s_wu[i] = __expf(ub[i] - umax);
  __syncthreads();

  float expalpha = stats[96];
  float norm = stats[b];

  if (blockIdx.y == 4) {  // dustbin column
    float ls = 0.f;
    for (int i = t; i < OT_P; i += 256) ls += s_wu[i];
    float S = blockSum(ls, sred);
    if (t == 0) {
      float alpha = stats[97];
      v[b * OT_P + OT_N] = stats[64 + b] - alpha - __logf(fmaxf(S, 1e-38f)) - umax;
    }
    return;
  }

  int c = blockIdx.y * 256 + t;
  const _Float16* Ec = E + ((size_t)b << 20) + c;
  float acc = 0.f;
#pragma unroll 8
  for (int i = 0; i < OT_M; ++i) acc += (float)Ec[(size_t)i << 10] * s_wu[i];
  acc += expalpha * s_wu[OT_N];
  float lnu = cm[b * OT_N + c] ? norm : NINF;
  v[b * OT_P + c] = lnu - __logf(fmaxf(acc, 1e-38f)) - umax;
}

// ---------------- K5: out = ps + u + v - norm ------------------------------
__global__ __launch_bounds__(256) void ot_final(
    const float* __restrict__ scores, const unsigned char* __restrict__ rm,
    const unsigned char* __restrict__ cm, const float* __restrict__ stats,
    const float* __restrict__ u, const float* __restrict__ v,
    float* __restrict__ out) {
  int j = blockIdx.x * 256 + threadIdx.x;
  if (j >= OT_P) return;
  int i = blockIdx.y, b = blockIdx.z;
  float norm = stats[b], alpha = stats[97];
  float ps;
  if (i < OT_M && j < OT_N) {
    ps = scores[((size_t)b << 20) + ((size_t)i << 10) + j];
    if (!rm[b * OT_M + i] || !cm[b * OT_N + j]) ps = NINF;
  } else {
    ps = alpha;
    if (i < OT_M && !rm[b * OT_M + i]) ps = NINF;
    if (j < OT_N && !cm[b * OT_N + j]) ps = NINF;
  }
  out[(size_t)b * (OT_P * OT_P) + (size_t)i * OT_P + j] =
      ps + u[b * OT_P + i] + v[b * OT_P + j] - norm;
}

// ---------------- host launcher --------------------------------------------
extern "C" void kernel_launch(void* const* d_in, const int* in_sizes, int n_in,
                              void* d_out, int out_size, void* d_ws,
                              size_t ws_size, hipStream_t stream) {
  const float* scores = (const float*)d_in[0];
  const float* alpha  = (const float*)d_in[1];
  const unsigned char* rm = (const unsigned char*)d_in[2];
  const unsigned char* cm = (const unsigned char*)d_in[3];
  float* out = (float*)d_out;

  // E (f16, 67 MB) aliased into d_out; only ot_final overwrites d_out, and it
  // never reads E. Small arrays (~263 KB) in d_ws.
  _Float16* E = (_Float16*)d_out;
  float* wsf   = (float*)d_ws;
  float* stats = wsf;                  // 128 floats
  float* u     = wsf + 128;            // B * 1025
  float* v     = u + OT_B * OT_P;      // B * 1025

  ot_stats_init<<<OT_B, 256, 0, stream>>>(rm, cm, alpha, stats, u, v);
  ot_exp<<<(unsigned)((size_t)OT_B * OT_MN / 1024), 256, 0, stream>>>(scores, rm, cm, E);
  for (int it = 0; it < OT_ITER; ++it) {
    ot_pass_u<<<dim3(OT_B, 9), 256, 0, stream>>>(E, v, u, stats, rm);
    ot_pass_v<<<dim3(OT_B, 5), 256, 0, stream>>>(E, u, v, stats, cm);
  }
  ot_final<<<dim3((OT_P + 255) / 256, OT_P, OT_B), 256, 0, stream>>>(
      scores, rm, cm, stats, u, v, out);
}
